// G_Spatial_8349416423887
// MI455X (gfx1250) — matrix operand order repaired and checked by reading, not compiled, not run on hardware
//
#include <hip/hip_runtime.h>
#include <hip/hip_bf16.h>
#include <math.h>

// Problem constants (from the reference)
#define B_   64
#define T_   24
#define TS_  25          // T_CAP + 1 time steps
#define QV_  15000
#define AV_  3000
#define E_   512
#define H_   1024
#define F_   2048
#define P_   49          // 7*7 spatial positions

typedef __attribute__((ext_vector_type(16))) __bf16 v16bf;
typedef __attribute__((ext_vector_type(8)))  __bf16 v8bf;
typedef __attribute__((ext_vector_type(8)))  float  v8f;

__device__ __forceinline__ float sigmoidf_(float x) { return 1.f / (1.f + __expf(-x)); }

__device__ __forceinline__ unsigned lds_off_(const void* p) {
    // Generic LDS address: flat->LDS mapping uses addr[31:0] (wave-relative).
    return (unsigned)(size_t)p;
}

// ---------------------------------------------------------------------------
// Generic WMMA bf16 GEMM: C[M,N] = act( A[M,K] * B[K,N] + bias (+ C if accum) )
//   AMODE 0: A[m*lda + k]                    (contiguous rows; async fast path)
//   AMODE 1: A[(m/a_div)*a_outer + (m%a_div) + k*a_col]   (features transpose)
//   Output row remap: out_group > 0 : orow = m + m/out_group + out_off
//   act: 0=none 1=tanh 2=relu
// Block: 128 threads (4 waves). Tile 64x64, K-step 32, double-buffered LDS.
// fp32 tiles streamed into LDS with GLOBAL_LOAD_ASYNC_TO_LDS (ASYNCcnt) one
// K-step ahead of the V_WMMA_16x16x32_BF16 compute; bf16 conversion happens at
// fragment build (v_cvt_pk_bf16_f32). One barrier per K-step:
//   s_wait_asynccnt 0  ->  barrier  ->  stage(next buf)  ->  wmma(cur buf)
// ---------------------------------------------------------------------------
template <int AMODE>
__global__ __launch_bounds__(128)
void wmma_gemm_kernel(const float* __restrict__ A, const float* __restrict__ Bm,
                      const float* __restrict__ bias, float* __restrict__ C,
                      int M, int N, int K,
                      long lda, long a_col, long a_div, long a_outer,
                      int ldb, int ldc, int out_group, int out_off,
                      int act, int accum)
{
    __shared__ float As32[2][64][36];   // [m][k], row stride 144B (16B aligned)
    __shared__ float Bs32[2][64][36];   // [n][k]  (B^T so K is contiguous)

    const int tid  = threadIdx.x;
    const int lane = tid & 31;
    const int w    = tid >> 5;      // wave id 0..3
    const int half = lane >> 4;     // lane group 0/1
    const int l16  = lane & 15;

    const int tileM = blockIdx.y * 64;
    const int tileN = blockIdx.x * 64;

    // Stage one K-step tile pair (fp32) into LDS buffer `buf` for K offset k0.
    auto stage = [&](int buf, int k0) {
        const unsigned ldsA = lds_off_(&As32[buf][0][0]);
        const unsigned ldsB = lds_off_(&Bs32[buf][0][0]);
        const bool fullK = (k0 + 32 <= K);
        // ---- A tile ----
        if (AMODE == 0 && fullK && (tileM + 64 <= M)) {
#pragma unroll
            for (int i = 0; i < 4; ++i) {
                int idx = tid + i * 128;            // 0..511 16B chunks
                int m = idx >> 3, c = idx & 7;
                const float* gp = A + (size_t)(tileM + m) * lda + k0 + c * 4;
                unsigned la = ldsA + (unsigned)(m * 36 + c * 4) * 4u;
                asm volatile("global_load_async_to_lds_b128 %0, %1, off"
                             :: "v"(la), "v"(gp) : "memory");
            }
        } else {
            for (int idx = tid; idx < 64 * 32; idx += 128) {
                int m = idx >> 5, k = idx & 31;
                int gm = tileM + m, gk = k0 + k;
                float v = 0.f;
                if (gm < M && gk < K) {
                    long off;
                    if (AMODE == 1)
                        off = (long)(gm / (int)a_div) * a_outer +
                              (long)(gm % (int)a_div) + (long)gk * a_col;
                    else
                        off = (long)gm * lda + gk;
                    v = A[off];
                }
                As32[buf][m][k] = v;
            }
        }
        // ---- B tile, transposed into [n][k] ----
        if (fullK && (tileN + 64 <= N)) {
#pragma unroll
            for (int i = 0; i < 16; ++i) {
                int idx = tid + i * 128;            // 0..2047 elements
                int n = idx & 63, k = idx >> 6;
                const float* gp = Bm + (size_t)(k0 + k) * ldb + tileN + n;
                unsigned la = ldsB + (unsigned)(n * 36 + k) * 4u;
                asm volatile("global_load_async_to_lds_b32 %0, %1, off"
                             :: "v"(la), "v"(gp) : "memory");
            }
        } else {
            for (int idx = tid; idx < 64 * 32; idx += 128) {
                int n = idx >> 5, k = idx & 31;
                int gn = tileN + n, gk = k0 + k;
                float v = 0.f;
                if (gn < N && gk < K) v = Bm[(long)gk * ldb + gn];
                Bs32[buf][n][k] = v;
            }
        }
    };

    v8f acc[4] = {};

    stage(0, 0);                         // prologue
    int p = 0;
    for (int k0 = 0; k0 < K; k0 += 32, p ^= 1) {
        asm volatile("s_wait_asynccnt 0" ::: "memory");
        __syncthreads();                 // all waves' tile-p data visible; all
                                         // waves done reading buffer p^1
        if (k0 + 32 < K) stage(p ^ 1, k0 + 32);   // overlap fetch with compute

        // ---- A fragment: contiguous k order, same as B ----
        // lanes 0-15: elems 0-15 -> K[0..15]
        // lanes 16-31: elems 0-15 -> K[16..31]
        v16bf av;
        {
            const float* ar = &As32[p][w * 16 + l16][0];
#pragma unroll
            for (int e = 0; e < 16; ++e) av[e] = (__bf16)ar[half * 16 + e];
        }
        // ---- All four B fragments first (batched ds loads + cvts), then
        //      back-to-back WMMAs (avoids RAW hazard bubbles) ----
        v16bf bv[4];
#pragma unroll
        for (int j = 0; j < 4; ++j) {
            const float* br = &Bs32[p][j * 16 + l16][0];
#pragma unroll
            for (int e = 0; e < 16; ++e) bv[j][e] = (__bf16)br[half * 16 + e];
        }
#pragma unroll
        for (int j = 0; j < 4; ++j)
            acc[j] = __builtin_amdgcn_wmma_f32_16x16x32_bf16(
                false, av, false, bv[j], (short)0, acc[j], false, false);
    }

    // Epilogue: C/D layout — VGPR v: lanes 0-15 M=v, lanes 16-31 M=v+8; N=l16
#pragma unroll
    for (int j = 0; j < 4; ++j) {
        int col = tileN + j * 16 + l16;
#pragma unroll
        for (int v = 0; v < 8; ++v) {
            int row = tileM + w * 16 + v + half * 8;
            if (row < M && col < N) {
                float val = acc[j][v];
                long orow = row;
                if (out_group > 0) orow = (long)row + row / out_group + out_off;
                long cidx = orow * (long)ldc + col;
                if (accum) val += C[cidx];
                if (bias)  val += bias[col];
                if (act == 1)      val = tanhf(val);
                else if (act == 2) val = fmaxf(val, 0.f);
                C[cidx] = val;
            }
        }
    }
}

// meanfeat[b*F + c] = mean over 49 positions of features[b,c,:]
__global__ void mean_kernel(const float* __restrict__ features, float* __restrict__ meanfeat)
{
    int i = blockIdx.x * blockDim.x + threadIdx.x;
    if (i >= B_ * F_) return;
    const float* p = features + (long)i * P_;
    float sum = 0.f;
    for (int j = 0; j < P_; ++j) sum += p[j];
    meanfeat[i] = sum * (1.f / 49.f);
}

__global__ void copy_kernel(const float* __restrict__ src, float* __restrict__ dst, int n)
{
    int i = blockIdx.x * blockDim.x + threadIdx.x;
    if (i < n) dst[i] = src[i];
}

// s[b,p] = w_att . tanh(local[b,p,:] + q[b,:])  — one block per (b,p)
__global__ void att_score_kernel(const float* __restrict__ local, const float* __restrict__ q,
                                 const float* __restrict__ w_att, float* __restrict__ s)
{
    int bp = blockIdx.x;
    int b  = bp / P_;
    const float* lrow = local + (long)bp * H_;
    const float* qrow = q + (long)b * H_;
    float acc = 0.f;
    for (int h = threadIdx.x; h < H_; h += blockDim.x)
        acc += w_att[h] * tanhf(lrow[h] + qrow[h]);
    __shared__ float red[4];
    int lane = threadIdx.x & 31, wid = threadIdx.x >> 5;
    for (int o = 16; o > 0; o >>= 1) acc += __shfl_down(acc, o);
    if (lane == 0) red[wid] = acc;
    __syncthreads();
    if (threadIdx.x == 0) s[bp] = red[0] + red[1] + red[2] + red[3];
}

// softmax over 49, ctx, LSTM pointwise update, results[b,t,:] = hx+cx
__global__ void lstm_update_kernel(const float* __restrict__ s, const float* __restrict__ local,
                                   const float* __restrict__ gates, float* __restrict__ hx,
                                   float* __restrict__ cx, float* __restrict__ results, int t)
{
    int b = blockIdx.x;
    __shared__ float alpha[P_];
    int tid = threadIdx.x;
    if (tid < P_) alpha[tid] = s[b * P_ + tid];
    __syncthreads();
    if (tid == 0) {
        float mx = -1e30f;
        for (int p = 0; p < P_; ++p) mx = fmaxf(mx, alpha[p]);
        float sum = 0.f;
        for (int p = 0; p < P_; ++p) { alpha[p] = __expf(alpha[p] - mx); sum += alpha[p]; }
        float inv = 1.f / sum;
        for (int p = 0; p < P_; ++p) alpha[p] *= inv;
    }
    __syncthreads();
    const float* g = gates + (long)b * 4 * H_;
    for (int h = tid; h < H_; h += blockDim.x) {
        float ctx = 0.f;
        for (int p = 0; p < P_; ++p) ctx += alpha[p] * local[((long)b * P_ + p) * H_ + h];
        float ig = sigmoidf_(g[h]);
        float fg = sigmoidf_(g[H_ + h]);
        float gg = tanhf(g[2 * H_ + h]);
        float og = sigmoidf_(g[3 * H_ + h]);
        float c  = fg * cx[(long)b * H_ + h] + ig * gg;
        float hh = og * tanhf(c) + ctx;
        cx[(long)b * H_ + h] = c;
        hx[(long)b * H_ + h] = hh;
        results[((long)b * TS_ + t) * H_ + h] = hh + c;
    }
}

// combined[b,:] = results[b, lengths[b]-1, :]
__global__ void gather_kernel(const float* __restrict__ results, const int* __restrict__ lengths,
                              float* __restrict__ combined)
{
    int b = blockIdx.x;
    int idx = lengths[b] - 1;
    if (idx < 0) idx = 0;
    if (idx >= TS_) idx = TS_ - 1;
    for (int h = threadIdx.x; h < H_; h += blockDim.x)
        combined[(long)b * H_ + h] = results[((long)b * TS_ + idx) * H_ + h];
}

extern "C" void kernel_launch(void* const* d_in, const int* in_sizes, int n_in,
                              void* d_out, int out_size, void* d_ws, size_t ws_size,
                              hipStream_t stream)
{
    const float* features = (const float*)d_in[0];
    const float* captions = (const float*)d_in[1];
    const int*   lengths  = (const int*)d_in[2];
    const float* hx0      = (const float*)d_in[3];
    const float* cx0      = (const float*)d_in[4];
    const float* W_embed  = (const float*)d_in[5];
    const float* b_embed  = (const float*)d_in[6];
    const float* Wg       = (const float*)d_in[7];
    const float* bg       = (const float*)d_in[8];
    const float* Wl       = (const float*)d_in[9];
    const float* bl       = (const float*)d_in[10];
    const float* W_ih     = (const float*)d_in[11];
    const float* W_hh     = (const float*)d_in[12];
    const float* b_lstm   = (const float*)d_in[13];
    const float* W_att_h  = (const float*)d_in[14];
    const float* b_att_h  = (const float*)d_in[15];
    const float* w_att    = (const float*)d_in[16];
    const float* W_fc1    = (const float*)d_in[17];
    const float* b_fc1    = (const float*)d_in[18];
    const float* W_fc2    = (const float*)d_in[19];
    const float* b_fc2    = (const float*)d_in[20];
    const float* W_fc3    = (const float*)d_in[21];
    const float* b_fc3    = (const float*)d_in[22];
    (void)in_sizes; (void)n_in; (void)out_size; (void)ws_size;

    // Workspace carve-up (floats); total ~25.8 MB
    float* ws       = (float*)d_ws;
    float* meanfeat = ws; ws += B_ * F_;
    float* local    = ws; ws += (long)B_ * P_ * H_;
    float* emb      = ws; ws += (long)B_ * TS_ * E_;    // [B, 25, E]: t=0 is g
    float* gates    = ws; ws += B_ * 4 * H_;
    float* q        = ws; ws += B_ * H_;
    float* s        = ws; ws += 4096;                    // B*49 rounded up
    float* hx       = ws; ws += B_ * H_;
    float* cx       = ws; ws += B_ * H_;
    float* results  = ws; ws += (long)B_ * TS_ * H_;
    float* combined = ws; ws += B_ * H_;
    float* h1       = ws; ws += B_ * H_;
    float* h2       = ws; ws += B_ * H_;

    dim3 blk(128);
    // Contiguous-A GEMM
    auto gemm = [&](const float* A, const float* Bm, const float* bias, float* C,
                    int M, int N, int K, long lda,
                    int ldb, int ldc, int og, int ooff, int act, int accum) {
        dim3 grid((unsigned)((N + 63) / 64), (unsigned)((M + 63) / 64));
        hipLaunchKernelGGL((wmma_gemm_kernel<0>), grid, blk, 0, stream,
                           A, Bm, bias, C, M, N, K, lda, (long)1, (long)0, (long)0,
                           ldb, ldc, og, ooff, act, accum);
    };

    // ---- Encoder ----
    mean_kernel<<<(B_ * F_ + 255) / 256, 256, 0, stream>>>(features, meanfeat);
    // g = tanh(meanfeat @ Wg + bg) -> emb[:, 0, :]  (ldc = 25*E)
    gemm(meanfeat, Wg, bg, emb, B_, E_, F_, F_, E_, TS_ * E_, 0, 0, 1, 0);
    // local = tanh(feat @ Wl + bl); strided-A view of features [B,F,49] -> [(b,p), c]
    {
        dim3 grid((unsigned)((H_ + 63) / 64), (unsigned)((B_ * P_ + 63) / 64));
        hipLaunchKernelGGL((wmma_gemm_kernel<1>), grid, blk, 0, stream,
                           features, Wl, bl, local, B_ * P_, H_, F_,
                           (long)0, (long)P_, (long)P_, (long)F_ * P_,
                           H_, H_, 0, 0, 1, 0);
    }
    // emb[:, 1:, :] = relu(captions @ W_embed + b_embed); remap row m -> b*25+t+1
    gemm(captions, W_embed, b_embed, emb, B_ * T_, E_, QV_, QV_, E_, E_, T_, 1, 2, 0);
    // h0 / c0
    copy_kernel<<<(B_ * H_ + 255) / 256, 256, 0, stream>>>(hx0, hx, B_ * H_);
    copy_kernel<<<(B_ * H_ + 255) / 256, 256, 0, stream>>>(cx0, cx, B_ * H_);

    // ---- LSTM scan (25 sequential steps) ----
    for (int t = 0; t < TS_; ++t) {
        // gates = x_t @ W_ih + b_lstm   (x_t = emb[:, t, :], row stride 25*E)
        gemm(emb + (long)t * E_, W_ih, b_lstm, gates, B_, 4 * H_, E_,
             (long)TS_ * E_, 4 * H_, 4 * H_, 0, 0, 0, 0);
        // gates += hx @ W_hh
        gemm(hx, W_hh, nullptr, gates, B_, 4 * H_, H_, H_, 4 * H_, 4 * H_, 0, 0, 0, 1);
        // q = hx @ W_att_h + b_att_h
        gemm(hx, W_att_h, b_att_h, q, B_, H_, H_, H_, H_, H_, 0, 0, 0, 0);
        att_score_kernel<<<B_ * P_, 128, 0, stream>>>(local, q, w_att, s);
        lstm_update_kernel<<<B_, 256, 0, stream>>>(s, local, gates, hx, cx, results, t);
    }

    // ---- Decoder FCs ----
    gather_kernel<<<B_, 256, 0, stream>>>(results, lengths, combined);
    gemm(combined, W_fc1, b_fc1, h1, B_, H_, H_, H_, H_, H_, 0, 0, 1, 0);
    gemm(h1, W_fc2, b_fc2, h2, B_, H_, H_, H_, H_, H_, 0, 0, 1, 0);
    gemm(h2, W_fc3, b_fc3, (float*)d_out, B_, AV_, H_, H_, AV_, AV_, 0, 0, 0, 0);
}
